// GDNBlock_56985626084082
// MI455X (gfx1250) — matrix operand order repaired and checked by reading, not compiled, not run on hardware
//
#include <hip/hip_runtime.h>

typedef __bf16 bf16;
typedef __attribute__((ext_vector_type(8)))  bf16  v8bf;
typedef __attribute__((ext_vector_type(16))) bf16  v16bf;
typedef __attribute__((ext_vector_type(8)))  float v8f;
typedef __attribute__((ext_vector_type(4)))  unsigned int u32x4;
typedef __attribute__((ext_vector_type(4)))  int   i32x4;
typedef __attribute__((ext_vector_type(8)))  int   i32x8;

#define DIM     1024
#define SEQ     4096
#define BATCH   2
#define ROWS    (BATCH*SEQ)     // 8192
#define HDIM    64
#define CHUNK   64
#define NCHUNK  (SEQ/CHUNK)     // 64
#define NCH     (BATCH*DIM)     // 2048 scan channels
#define RES_SCALE 0.70710678118654752440f
#define EPS1    1e-6f
#define EPSQ    1.1920928955078125e-07f  // FLT_EPSILON

#if __has_builtin(__builtin_amdgcn_tensor_load_to_lds)
#define USE_TDM 1
#else
#define USE_TDM 0
#endif

static __device__ inline v16bf concat8(v8bf lo, v8bf hi) {
  return __builtin_shufflevector(lo, hi, 0,1,2,3,4,5,6,7,8,9,10,11,12,13,14,15);
}

#if USE_TDM
// ---------------------------------------------------------------------------
// TDM: DMA one 128-row x 32-col bf16 tile (row stride K elems) into LDS with
// hardware padding: rows of 64B followed by 16B pad -> 80B LDS row stride,
// matching the fragment-gather layout below.
// ---------------------------------------------------------------------------
static __device__ inline void tdm_load_tile(unsigned lds_addr,
                                            unsigned long long gaddr,
                                            int K) {
  u32x4 g0;
  g0[0] = 1u;                                            // count=1, user mode
  g0[1] = lds_addr;                                      // LDS byte offset
  g0[2] = (unsigned)(gaddr & 0xFFFFFFFFull);             // global_addr[31:0]
  g0[3] = (unsigned)((gaddr >> 32) & 0x01FFFFFFull)      // global_addr[56:32]
        | 0x80000000u;                                   // type=2 ("image")
  const unsigned td0 = (unsigned)K;                      // tensor_dim0 (elems)
  const unsigned td1 = 128u;                             // tensor_dim1 (rows)
  i32x8 g1;
  g1[0] = (int)((1u << 16)        // data_size = 2 bytes
              | (1u << 20)        // pad_enable
              | (3u << 22)        // pad_interval: 16 DWORDs (one 64B row)
              | (3u << 25));      // pad_amount: 4 DWORDs (16B)
  g1[1] = (int)((td0 & 0xFFFFu) << 16);                  // tensor_dim0 lo16
  g1[2] = (int)((td0 >> 16) | ((td1 & 0xFFFFu) << 16));  // td0 hi / td1 lo
  g1[3] = (int)((td1 >> 16) | (32u << 16));              // td1 hi / tile_dim0=32
  g1[4] = (int)128u;                                     // tile_dim1=128, dim2=0
  g1[5] = (int)td0;                                      // dim0 stride lo (elems)
  g1[6] = 0;                                             // stride hi / dim1 stride
  g1[7] = 0;
  i32x4 z4 = {0, 0, 0, 0};
#if defined(__clang_major__) && __clang_major__ >= 23
  i32x8 z8 = {0, 0, 0, 0, 0, 0, 0, 0};
  __builtin_amdgcn_tensor_load_to_lds(g0, g1, z4, z4, z8, 0);
#else
  __builtin_amdgcn_tensor_load_to_lds(g0, g1, z4, z4, 0);
#endif
}
#endif

// ---------------------------------------------------------------------------
// fp32 -> bf16 conversion (weights)
// ---------------------------------------------------------------------------
__global__ void cvt_bf16(const float* __restrict__ src, bf16* __restrict__ dst, int n) {
  int i = blockIdx.x * blockDim.x + threadIdx.x;
  int stride = gridDim.x * blockDim.x;
  for (; i < n; i += stride) dst[i] = (bf16)src[i];
}

// ---------------------------------------------------------------------------
// RMSNorm over rows of 1024, scaled by weight, output bf16
// ---------------------------------------------------------------------------
__global__ __launch_bounds__(256) void rmsnorm_bf16_kernel(
    const float* __restrict__ x, const float* __restrict__ w,
    bf16* __restrict__ out, float eps)
{
  const int row = blockIdx.x;
  const int t   = threadIdx.x;
  const float4 v = ((const float4*)(x + (size_t)row * DIM))[t];
  float ss = v.x*v.x + v.y*v.y + v.z*v.z + v.w*v.w;
  #pragma unroll
  for (int off = 16; off > 0; off >>= 1) ss += __shfl_xor(ss, off, 32);
  __shared__ float wsum[8];
  if ((t & 31) == 0) wsum[t >> 5] = ss;
  __syncthreads();
  float tot = 0.f;
  #pragma unroll
  for (int i = 0; i < 8; i++) tot += wsum[i];
  const float sc = rsqrtf(tot * (1.f / DIM) + eps);
  const int c = t * 4;
  bf16* o = out + (size_t)row * DIM + c;
  o[0] = (bf16)(v.x * sc * w[c + 0]);
  o[1] = (bf16)(v.y * sc * w[c + 1]);
  o[2] = (bf16)(v.z * sc * w[c + 2]);
  o[3] = (bf16)(v.w * sc * w[c + 3]);
}

// ---------------------------------------------------------------------------
// bf16 WMMA GEMM: C[M,N] = A[M,K] @ B[N,K]^T, f32 accumulate.
// 256 threads = 8 waves, 128x128 block tile, K-step 32.
// Tiles staged into LDS by the Tensor Data Mover (TENSORcnt-tracked),
// double-buffered; wave0 issues DMA + waits, workgroup barrier publishes.
// EPI: 0 = store f32, 1 = f32 store of Res + scale*acc, 2 = silu -> bf16
// ---------------------------------------------------------------------------
template <int EPI>
__global__ __launch_bounds__(256) void gemm_bf16(
    const bf16* __restrict__ A, const bf16* __restrict__ B,
    const float* __restrict__ Res, void* __restrict__ C,
    int M, int N, int K, float scale)
{
  __shared__ __align__(16) bf16 As[2][128 * 40];
  __shared__ __align__(16) bf16 Bs[2][128 * 40];

  const int tid  = threadIdx.x;
  const int lane = tid & 31;
  const int wave = tid >> 5;
  const int wm = wave >> 2;            // 0..1 : 64 rows each
  const int wn = wave & 3;             // 0..3 : 32 cols each
  const int m0 = blockIdx.y * 128;
  const int n0 = blockIdx.x * 128;

  // fragment gather bases (per doc'd 16-bit A / B lane layouts), 80B LDS rows
  const int aRowB = wm * 64 + (lane & 15);
  const int aKh   = (lane >> 4) * 8;                 // K halves 0..7 / 8..15 (+16)
  const int bRowB = wn * 32 + (lane & 15);
  const int bKh   = (lane >> 4) * 8;                 // K halves 0..7 / 8..15 (+16)

  v8f acc[4][2] = {};
  const int kTiles = K >> 5;

#if USE_TDM
  const unsigned ldsA = (unsigned)(uintptr_t)(&As[0][0]);
  const unsigned ldsB = (unsigned)(uintptr_t)(&Bs[0][0]);
  const unsigned long long gA =
      (unsigned long long)(uintptr_t)A + (unsigned long long)((size_t)m0 * K) * 2ull;
  const unsigned long long gB =
      (unsigned long long)(uintptr_t)B + (unsigned long long)((size_t)n0 * K) * 2ull;

  if (wave == 0) {                     // prologue: tile 0 via TDM
    tdm_load_tile(ldsA, gA, K);
    tdm_load_tile(ldsB, gB, K);
    __builtin_amdgcn_s_wait_tensorcnt(0);
  }
  __syncthreads();
#else
  const int ldr     = tid >> 1;                      // 0..127
  const int lds_off = ldr * 40 + (tid & 1) * 16;
  const size_t aG = (size_t)(m0 + ldr) * K + (tid & 1) * 16;
  const size_t bG = (size_t)(n0 + ldr) * K + (tid & 1) * 16;
  *(v8bf*)&As[0][lds_off]     = *(const v8bf*)(A + aG);
  *(v8bf*)&As[0][lds_off + 8] = *(const v8bf*)(A + aG + 8);
  *(v8bf*)&Bs[0][lds_off]     = *(const v8bf*)(B + bG);
  *(v8bf*)&Bs[0][lds_off + 8] = *(const v8bf*)(B + bG + 8);
  __syncthreads();
#endif

  for (int kt = 0; kt < kTiles; ++kt) {
    const int cur = kt & 1, nxt = cur ^ 1;

#if USE_TDM
    if (wave == 0 && kt + 1 < kTiles) {  // DMA next K-tile into other buffer
      const unsigned long long koff = (unsigned long long)(kt + 1) * 64ull;
      tdm_load_tile(ldsA + (unsigned)nxt * 10240u, gA + koff, K);
      tdm_load_tile(ldsB + (unsigned)nxt * 10240u, gB + koff, K);
    }
#else
    v8bf pa0, pa1, pb0, pb1;
    if (kt + 1 < kTiles) {
      const size_t ka = aG + (size_t)(kt + 1) * 32;
      const size_t kb = bG + (size_t)(kt + 1) * 32;
      pa0 = *(const v8bf*)(A + ka);
      pa1 = *(const v8bf*)(A + ka + 8);
      pb0 = *(const v8bf*)(B + kb);
      pb1 = *(const v8bf*)(B + kb + 8);
    }
#endif

    v16bf afr[4], bfr[2];
    #pragma unroll
    for (int mt = 0; mt < 4; ++mt) {
      const bf16* p = &As[cur][(aRowB + mt * 16) * 40 + aKh];
      afr[mt] = concat8(*(const v8bf*)p, *(const v8bf*)(p + 16));
    }
    #pragma unroll
    for (int nt = 0; nt < 2; ++nt) {
      const bf16* p = &Bs[cur][(bRowB + nt * 16) * 40 + bKh];
      bfr[nt] = concat8(*(const v8bf*)p, *(const v8bf*)(p + 16));
    }
    #pragma unroll
    for (int mt = 0; mt < 4; ++mt)
      #pragma unroll
      for (int nt = 0; nt < 2; ++nt)
        acc[mt][nt] = __builtin_amdgcn_wmma_f32_16x16x32_bf16(
            false, afr[mt], false, bfr[nt], (short)0, acc[mt][nt], false, false);

#if USE_TDM
    if (wave == 0 && kt + 1 < kTiles)
      __builtin_amdgcn_s_wait_tensorcnt(0);
#else
    if (kt + 1 < kTiles) {
      *(v8bf*)&As[nxt][lds_off]     = pa0;
      *(v8bf*)&As[nxt][lds_off + 8] = pa1;
      *(v8bf*)&Bs[nxt][lds_off]     = pb0;
      *(v8bf*)&Bs[nxt][lds_off + 8] = pb1;
    }
#endif
    __syncthreads();
  }

  // epilogue: C/D layout -> lanes 0-15: M=r, lanes 16-31: M=r+8
  #pragma unroll
  for (int mt = 0; mt < 4; ++mt)
    #pragma unroll
    for (int nt = 0; nt < 2; ++nt)
      #pragma unroll
      for (int r = 0; r < 8; ++r) {
        const int row = m0 + wm * 64 + mt * 16 + ((lane >> 4) << 3) + r;
        const int col = n0 + wn * 32 + nt * 16 + (lane & 15);
        const size_t idx = (size_t)row * N + col;
        const float v = acc[mt][nt][r];
        if constexpr (EPI == 0) {
          ((float*)C)[idx] = v;
        } else if constexpr (EPI == 1) {
          ((float*)C)[idx] = Res[idx] + scale * v;
        } else {
          ((bf16*)C)[idx] = (bf16)(v / (1.f + __expf(-v)));
        }
      }
}

// ---------------------------------------------------------------------------
// Chunk-parallel scan of s_t = a_t*s_{t-1} + u_t (reference's clamped form)
// proj row layout: [q | k | v | a | g] each DIM wide
// ---------------------------------------------------------------------------
static __device__ inline void load_au(const float* __restrict__ proj,
                                      int row, int col, float& a, float& u) {
  const size_t off = (size_t)row * (5 * DIM);
  const float pk = proj[off + DIM     + col];
  const float pv = proj[off + 2*DIM   + col];
  const float pa = proj[off + 3*DIM   + col];
  float av = 1.f / (1.f + __expf(-(pa + 2.f)));
  a = fminf(fmaxf(av, 0.6f), 0.9995f);
  u = tanhf(pk) * pv;
}

__global__ __launch_bounds__(256) void scan_phaseA(
    const float* __restrict__ proj, float* __restrict__ P, float* __restrict__ S)
{
  const int gid = blockIdx.x * blockDim.x + threadIdx.x;  // gid = c*NCH + ch
  const int ch  = gid & (NCH - 1);
  const int c   = gid >> 11;
  const int b   = ch >> 10;
  const int col = ch & (DIM - 1);
  const int row0 = b * SEQ + c * CHUNK;
  float prefix = 1.f, acc = 0.f;
  for (int t = 0; t < CHUNK; ++t) {
    float a, u;
    load_au(proj, row0 + t, col, a, u);
    prefix *= a;
    acc += u / fmaxf(prefix, 1e-6f);
  }
  P[gid] = prefix;
  S[gid] = prefix * acc;   // chunk-end state starting from 0
}

__global__ __launch_bounds__(256) void scan_phaseB(
    const float* __restrict__ P, const float* __restrict__ S, float* __restrict__ Sin)
{
  const int ch = blockIdx.x * blockDim.x + threadIdx.x;   // 2048 channels
  float st = 0.f;
  for (int c = 0; c < NCHUNK; ++c) {
    Sin[c * NCH + ch] = st;
    st = P[c * NCH + ch] * st + S[c * NCH + ch];
  }
}

__global__ __launch_bounds__(256) void scan_phaseC(
    const float* __restrict__ proj, const float* __restrict__ Sin,
    float* __restrict__ mem)
{
  const int gid = blockIdx.x * blockDim.x + threadIdx.x;
  const int ch  = gid & (NCH - 1);
  const int c   = gid >> 11;
  const int b   = ch >> 10;
  const int col = ch & (DIM - 1);
  const int row0 = b * SEQ + c * CHUNK;
  const float st = Sin[gid];
  float prefix = 1.f, acc = 0.f;
  for (int t = 0; t < CHUNK; ++t) {
    float a, u;
    load_au(proj, row0 + t, col, a, u);
    prefix *= a;
    acc += u / fmaxf(prefix, 1e-6f);
    mem[(size_t)(row0 + t) * DIM + col] = prefix * (st + acc);
  }
}

// ---------------------------------------------------------------------------
// y = sigmoid(g) * (rms(q)*mem) + (1-sigmoid(g)) * v   (per-head rms over 64)
// ---------------------------------------------------------------------------
__global__ __launch_bounds__(256) void y_kernel(
    const float* __restrict__ proj, const float* __restrict__ mem,
    bf16* __restrict__ y)
{
  const int row = blockIdx.x;
  const int t   = threadIdx.x;
  const int c   = t * 4;                       // 4 cols/thread; head = 16 threads
  const size_t po = (size_t)row * (5 * DIM);
  const float4 q  = *(const float4*)(proj + po + c);
  const float4 vv = *(const float4*)(proj + po + 2 * DIM + c);
  const float4 g  = *(const float4*)(proj + po + 4 * DIM + c);
  const float4 m  = *(const float4*)(mem + (size_t)row * DIM + c);

  float ss = q.x*q.x + q.y*q.y + q.z*q.z + q.w*q.w;
  #pragma unroll
  for (int off = 8; off > 0; off >>= 1) ss += __shfl_xor(ss, off, 16);
  const float rs = rsqrtf(ss * (1.f / HDIM) + EPSQ);

  bf16* o = y + (size_t)row * DIM + c;
  const float gx = 1.f / (1.f + __expf(-g.x));
  const float gy = 1.f / (1.f + __expf(-g.y));
  const float gz = 1.f / (1.f + __expf(-g.z));
  const float gw = 1.f / (1.f + __expf(-g.w));
  o[0] = (bf16)(gx * (q.x * rs * m.x) + (1.f - gx) * vv.x);
  o[1] = (bf16)(gy * (q.y * rs * m.y) + (1.f - gy) * vv.y);
  o[2] = (bf16)(gz * (q.z * rs * m.z) + (1.f - gz) * vv.z);
  o[3] = (bf16)(gw * (q.w * rs * m.w) + (1.f - gw) * vv.w);
}

// ---------------------------------------------------------------------------
extern "C" void kernel_launch(void* const* d_in, const int* in_sizes, int n_in,
                              void* d_out, int out_size, void* d_ws, size_t ws_size,
                              hipStream_t stream)
{
  const float* x      = (const float*)d_in[0];
  const float* n1w    = (const float*)d_in[1];
  const float* n2w    = (const float*)d_in[2];
  const float* w_in   = (const float*)d_in[3];   // [5120,1024]
  const float* w_out  = (const float*)d_in[4];   // [1024,1024]
  const float* w_fc   = (const float*)d_in[5];   // [4096,1024]
  const float* w_proj = (const float*)d_in[6];   // [1024,4096]
  float* out = (float*)d_out;

  char* p = (char*)d_ws;
  auto alloc = [&](size_t bytes) -> char* {
    char* r = p;
    p += (bytes + 255) & ~(size_t)255;
    return r;
  };
  bf16*  hb     = (bf16*)alloc((size_t)ROWS * DIM * 2);       // h1 then h2
  bf16*  yb     = (bf16*)alloc((size_t)ROWS * DIM * 2);       // mixer y
  bf16*  tb     = (bf16*)alloc((size_t)ROWS * 4096 * 2);      // silu(fc) hidden
  bf16*  winb   = (bf16*)alloc((size_t)5 * DIM * DIM * 2);
  bf16*  woutb  = (bf16*)alloc((size_t)DIM * DIM * 2);
  bf16*  wfcb   = (bf16*)alloc((size_t)4096 * DIM * 2);
  bf16*  wprojb = (bf16*)alloc((size_t)DIM * 4096 * 2);
  float* proj   = (float*)alloc((size_t)ROWS * 5 * DIM * 4);  // q,k,v,a,g
  float* x2     = (float*)alloc((size_t)ROWS * DIM * 4);      // x + rs*mixer
  float* mem    = (float*)alloc((size_t)ROWS * DIM * 4);      // scan states
  float* Pb     = (float*)alloc((size_t)NCHUNK * NCH * 4);
  float* Sb     = (float*)alloc((size_t)NCHUNK * NCH * 4);
  float* Sin    = (float*)alloc((size_t)NCHUNK * NCH * 4);
  (void)ws_size; (void)in_sizes; (void)n_in; (void)out_size;

  // weight conversion to bf16
  cvt_bf16<<<2048, 256, 0, stream>>>(w_in,   winb,   5 * DIM * DIM);
  cvt_bf16<<<1024, 256, 0, stream>>>(w_out,  woutb,  DIM * DIM);
  cvt_bf16<<<2048, 256, 0, stream>>>(w_fc,   wfcb,   4096 * DIM);
  cvt_bf16<<<2048, 256, 0, stream>>>(w_proj, wprojb, DIM * 4096);

  // h1 = norm1 * rms(x)
  rmsnorm_bf16_kernel<<<ROWS, 256, 0, stream>>>(x, n1w, hb, EPS1);

  // proj = h1 @ in_proj^T  [8192 x 5120]
  gemm_bf16<0><<<dim3(5 * DIM / 128, ROWS / 128), 256, 0, stream>>>(
      hb, winb, nullptr, proj, ROWS, 5 * DIM, DIM, 0.f);

  // chunk-parallel linear recurrence
  scan_phaseA<<<(NCHUNK * NCH) / 256, 256, 0, stream>>>(proj, Pb, Sb);
  scan_phaseB<<<NCH / 256, 256, 0, stream>>>(Pb, Sb, Sin);
  scan_phaseC<<<(NCHUNK * NCH) / 256, 256, 0, stream>>>(proj, Sin, mem);

  // gated combine
  y_kernel<<<ROWS, 256, 0, stream>>>(proj, mem, yb);

  // x2 = x + RES_SCALE * (y @ out_proj^T)
  gemm_bf16<1><<<dim3(DIM / 128, ROWS / 128), 256, 0, stream>>>(
      yb, woutb, x, x2, ROWS, DIM, DIM, RES_SCALE);

  // h2 = norm2 * rms(x2)
  rmsnorm_bf16_kernel<<<ROWS, 256, 0, stream>>>(x2, n2w, hb, EPS1);

  // t = silu(h2 @ fc^T)  [8192 x 4096] bf16
  gemm_bf16<2><<<dim3(4096 / 128, ROWS / 128), 256, 0, stream>>>(
      hb, wfcb, nullptr, tb, ROWS, 4096, DIM, 0.f);

  // out = x2 + RES_SCALE * (t @ proj_w^T)
  gemm_bf16<1><<<dim3(DIM / 128, ROWS / 128), 256, 0, stream>>>(
      tb, wprojb, x2, out, ROWS, DIM, 4096, RES_SCALE);
}
